// RelPartialLearnableMultiHeadAttn_9242769622563
// MI455X (gfx1250) — compile-verified, matrix operand order checked by reading, not run on hardware
//
#include <hip/hip_runtime.h>
#include <stdint.h>

typedef __bf16 bf16;
typedef __bf16 v16bf __attribute__((ext_vector_type(16)));
typedef float  v8f   __attribute__((ext_vector_type(8)));
typedef uint32_t v4u __attribute__((ext_vector_type(4)));
typedef uint32_t v8u __attribute__((ext_vector_type(8)));

struct __attribute__((aligned(16))) U128 { uint32_t w[4]; };
union FragA { U128 u[2]; v16bf v; };
static_assert(sizeof(FragA) == 32, "frag size");

#define NUM_B 4
#define SEQ_Q 2048
#define DIM   1024
#define NHEAD 16
#define DHEAD 64
#define ROWS  (NUM_B * SEQ_Q)   /* 8192 */
#define N_QKV (3 * DIM)         /* 3072 */

__device__ __forceinline__ v8f wmma_bf16(v16bf a, v16bf b, v8f c) {
  return __builtin_amdgcn_wmma_f32_16x16x32_bf16(false, a, false, b, (short)0, c,
                                                 false, false);
}
__device__ __forceinline__ U128 ldg16(const bf16* p) { return *(const U128*)p; }

// CDNA5 async DMA: global -> LDS, 16B per lane, tracked by ASYNCcnt.
__device__ __forceinline__ void async_ld_b128(uint32_t lds_off, const void* gptr) {
  asm volatile("global_load_async_to_lds_b128 %0, %1, off"
               :: "v"(lds_off), "v"(gptr) : "memory");
}

// CDNA5 Tensor Data Mover: 2D tile -> LDS via descriptor (D#), TENSORcnt-tracked.
// group0: count=1 | lds_addr | global_addr[56:0] | type=2 ("image")
// groups 2/3 zero (2-D tensor).
__device__ __forceinline__ void tdm_load_2d(uint32_t lds_off, const void* gptr,
                                            v8u g1) {
  uint64_t ga = (uint64_t)(uintptr_t)gptr;
  v4u g0;
  g0.x = 1u;                                 // count=1, user descriptor
  g0.y = lds_off;                            // lds_addr (bytes)
  g0.z = (uint32_t)ga;                       // global_addr[31:0]
  g0.w = (uint32_t)(ga >> 32) | (2u << 30);  // global_addr[56:32] | type=2
  v4u gz = {0u, 0u, 0u, 0u};
  asm volatile("tensor_load_to_lds %0, %1, %2, %3"
               :: "s"(g0), "s"(g1), "s"(gz), "s"(gz) : "memory");
}

// ---------------- f32 -> bf16 convert ----------------
__global__ __launch_bounds__(256) void cvt_bf16(const float* __restrict__ in,
                                                bf16* __restrict__ out, int n) {
  int i = blockIdx.x * 256 + threadIdx.x;
  if (i < n) out[i] = (bf16)in[i];
}

// ---------------- f32 (K,N) -> bf16 (N,K) transpose-convert ----------------
__global__ __launch_bounds__(256) void transpose_cvt_bf16(
    const float* __restrict__ in, bf16* __restrict__ out, int K, int N) {
  __shared__ float tile[32][33];
  const int n0 = blockIdx.x * 32, k0 = blockIdx.y * 32;
  const int tx = threadIdx.x & 31, ty = threadIdx.x >> 5;  // 32 x 8
#pragma unroll
  for (int r = 0; r < 32; r += 8)
    tile[ty + r][tx] = in[(size_t)(k0 + ty + r) * N + n0 + tx];
  __syncthreads();
#pragma unroll
  for (int r = 0; r < 32; r += 8)
    out[(size_t)(n0 + ty + r) * K + k0 + tx] = (bf16)tile[tx][ty + r];
}

// ---------------- tiled bf16 WMMA GEMM, block tile 64x128, 8 waves ----------------
// A: (M,K) bf16 row-major; Bt: (N,K) bf16 row-major (pre-transposed weights).
// Double-buffered staging: A tile via per-lane async b128 (ASYNCcnt),
// B tile via one TDM descriptor per stage (TENSORcnt) with HW pad -> stride-40 LDS.
// mode 0: QKV epilogue (q+rw_bias, q+rr_bias, k, V transposed per (b,h))
// mode 1: plain bf16 store    mode 2: f32 store
__global__ __launch_bounds__(256) void gemm_wmma_bf16(
    const bf16* __restrict__ A, const bf16* __restrict__ Bt,
    int M, int N, int K, int mode,
    bf16* __restrict__ out_qrw, bf16* __restrict__ out_qrr,
    bf16* __restrict__ out_k, bf16* __restrict__ out_v,
    const float* __restrict__ rwb, const float* __restrict__ rrb,
    bf16* __restrict__ out_bf, float* __restrict__ out_f32) {
  __shared__ __attribute__((aligned(16))) bf16 aT[2][64 * 40];
  __shared__ __attribute__((aligned(16))) bf16 bT[2][128 * 40];

  const int t = threadIdx.x;
  const int lane = t & 31;
  const int wave = t >> 5;
  const int wm = wave & 1;   // 2 waves along M
  const int wn = wave >> 1;  // 4 waves along N
  const int m0 = blockIdx.y * 64;
  const int n0 = blockIdx.x * 128;
  const int am = lane & 15;
  const int kh = (lane >> 4) * 8;

  // A staging: 64x32 halves -> exactly 1 b128 per thread
  const int a_row = t >> 2, a_ch = (t & 3) * 8;
  const bf16* aptr = A + (size_t)(m0 + a_row) * K + a_ch;

  // B tile TDM descriptor group1: data_size=2B; pad_enable, pad_interval=16DW,
  // pad_amount=4DW (row 64B + pad 16B = stride 40 halves); tile 32 x 128;
  // tensor dim0=K, dim1=N, dim0_stride=K (elements).
  v8u g1b;
  g1b[0] = (1u << 16) | (1u << 20) | (3u << 22) | (3u << 25);
  g1b[1] = ((uint32_t)K & 0xFFFFu) << 16;
  g1b[2] = ((uint32_t)K >> 16) | (((uint32_t)N & 0xFFFFu) << 16);
  g1b[3] = ((uint32_t)N >> 16) | (32u << 16);   // tile_dim0 = 32
  g1b[4] = 128u;                                 // tile_dim1 = 128
  g1b[5] = (uint32_t)K;                          // tensor_dim0_stride
  g1b[6] = 0u;
  g1b[7] = 0u;

  auto issue = [&](int k0, int buf) {
    uint32_t la = (uint32_t)(uintptr_t)&aT[buf][a_row * 40 + a_ch];
    async_ld_b128(la, aptr + k0);
    if (wave == 0) {
      uint32_t lb = (uint32_t)(uintptr_t)&bT[buf][0];
      tdm_load_2d(lb, Bt + (size_t)n0 * K + k0, g1b);
    }
  };

  v8f acc[2][2] = {};
  const int nk = K >> 5;
  issue(0, 0);

  for (int kk = 0; kk < nk; ++kk) {
    if (kk + 1 < nk) {
      issue((kk + 1) << 5, (kk + 1) & 1);
      asm volatile("s_wait_asynccnt 0x1" ::: "memory");   // stage kk A landed
      if (wave == 0) __builtin_amdgcn_s_wait_tensorcnt(1);  // stage kk B landed
    } else {
      asm volatile("s_wait_asynccnt 0x0" ::: "memory");
      if (wave == 0) __builtin_amdgcn_s_wait_tensorcnt(0);
    }
    __syncthreads();

    const bf16* ab = aT[kk & 1];
    const bf16* bb = bT[kk & 1];
    FragA fa[2], fb[2];
#pragma unroll
    for (int mt = 0; mt < 2; ++mt) {
      int row = wm * 32 + mt * 16 + am;
      fa[mt].u[0] = *(const U128*)&ab[row * 40 + kh];
      fa[mt].u[1] = *(const U128*)&ab[row * 40 + 16 + kh];
    }
#pragma unroll
    for (int nt = 0; nt < 2; ++nt) {
      int col = wn * 32 + nt * 16 + am;
      fb[nt].u[0] = *(const U128*)&bb[col * 40 + kh];
      fb[nt].u[1] = *(const U128*)&bb[col * 40 + 16 + kh];
    }
#pragma unroll
    for (int mt = 0; mt < 2; ++mt)
#pragma unroll
      for (int nt = 0; nt < 2; ++nt)
        acc[mt][nt] = wmma_bf16(fa[mt].v, fb[nt].v, acc[mt][nt]);
    __syncthreads();  // all waves done with buf (kk&1) before it is re-issued
  }

  // C layout: lane holds rows (lane/16)*8+v, col lane%16
#pragma unroll
  for (int mt = 0; mt < 2; ++mt)
#pragma unroll
    for (int nt = 0; nt < 2; ++nt)
#pragma unroll
      for (int v = 0; v < 8; ++v) {
        int r = m0 + wm * 32 + mt * 16 + (lane >> 4) * 8 + v;
        int c = n0 + wn * 32 + nt * 16 + am;
        float val = acc[mt][nt][v];
        if (mode == 0) {
          int sect = c >> 10, cc = c & 1023;
          if (sect == 0) {
            size_t o = (size_t)r * DIM + cc;
            out_qrw[o] = (bf16)(val + rwb[cc]);
            out_qrr[o] = (bf16)(val + rrb[cc]);
          } else if (sect == 1) {
            out_k[(size_t)r * DIM + cc] = (bf16)val;
          } else {
            // V stored transposed per (b,h): [(b*H+h)*Dh + d][key]
            int bb_ = r >> 11, key = r & (SEQ_Q - 1);
            int h = cc >> 6, d = cc & 63;
            out_v[((size_t)((bb_ * NHEAD + h) * DHEAD + d)) * SEQ_Q + key] =
                (bf16)val;
          }
        } else if (mode == 1) {
          out_bf[(size_t)r * N + c] = (bf16)val;
        } else {
          out_f32[(size_t)r * N + c] = val;
        }
      }
}

// ---------------- flash attention with rel-shift band (1 wave / 16 q rows) ----------------
__global__ __launch_bounds__(32) void attn_flash_wmma(
    const bf16* __restrict__ qrw, const bf16* __restrict__ qrr,
    const bf16* __restrict__ kbf, const bf16* __restrict__ vtr,
    const bf16* __restrict__ rkb, bf16* __restrict__ avec) {
  __shared__ __attribute__((aligned(16))) float bdl[16 * 48];  // BD band tile
  __shared__ __attribute__((aligned(16))) bf16 pl[16 * 40];    // P staging

  const int lane = threadIdx.x & 31;
  const int bh = blockIdx.y;
  const int b = bh >> 4;
  const int h = bh & 15;
  const int i0 = blockIdx.x * 16;
  const int m = lane & 15;
  const int kh = (lane >> 4) * 8;
  const int rbase = (lane >> 4) * 8;

  // preload q fragments (16 rows x 64 dims -> 2 K-step frags each)
  FragA aw0, aw1, ar0, ar1;
  {
    const bf16* q0 = qrw + ((size_t)(b * SEQ_Q + i0 + m) * DIM) + h * DHEAD;
    aw0.u[0] = ldg16(q0 + kh);      aw0.u[1] = ldg16(q0 + 16 + kh);
    aw1.u[0] = ldg16(q0 + 32 + kh); aw1.u[1] = ldg16(q0 + 48 + kh);
    const bf16* q1 = qrr + ((size_t)(b * SEQ_Q + i0 + m) * DIM) + h * DHEAD;
    ar0.u[0] = ldg16(q1 + kh);      ar0.u[1] = ldg16(q1 + 16 + kh);
    ar1.u[0] = ldg16(q1 + 32 + kh); ar1.u[1] = ldg16(q1 + 48 + kh);
  }

  v8f o0 = {}, o1 = {}, o2 = {}, o3 = {};
  float mrow[8], lrow[8];
#pragma unroll
  for (int v = 0; v < 8; ++v) { mrow[v] = -1.0e30f; lrow[v] = 0.0f; }
  const float scale = 0.125f;  // 1/sqrt(64)

  for (int j0 = 0; j0 < i0 + 16; j0 += 32) {
    // AC = (q + r_w_bias) . K^T  for 32 keys (B frags in AMD table k order over d)
    v8f s0 = {}, s1 = {};
#pragma unroll
    for (int nt = 0; nt < 2; ++nt) {
      const bf16* kb =
          kbf + ((size_t)(b * SEQ_Q + j0 + nt * 16 + m) * DIM) + h * DHEAD;
      FragA b0, b1;
      b0.u[0] = ldg16(kb + kh);      b0.u[1] = ldg16(kb + 16 + kh);
      b1.u[0] = ldg16(kb + 32 + kh); b1.u[1] = ldg16(kb + 48 + kh);
      v8f s = nt ? s1 : s0;
      s = wmma_bf16(aw0.v, b0.v, s);
      s = wmma_bf16(aw1.v, b1.v, s);
      if (nt) s1 = s; else s0 = s;
    }

    // BD band: shifted BD[i][j] = BD_orig[i][Q-1-i+j]; p window p0..p0+47
    const int p0 = SEQ_Q - 16 - i0 + j0;
#pragma unroll
    for (int nt = 0; nt < 3; ++nt) {
      int p = p0 + nt * 16 + m;
      p = p < 0 ? 0 : (p > SEQ_Q - 1 ? SEQ_Q - 1 : p);  // clamped == masked region
      const bf16* rb = rkb + (size_t)p * DIM + h * DHEAD;
      FragA b0, b1;
      b0.u[0] = ldg16(rb + kh);      b0.u[1] = ldg16(rb + 16 + kh);
      b1.u[0] = ldg16(rb + 32 + kh); b1.u[1] = ldg16(rb + 48 + kh);
      v8f bd = {};
      bd = wmma_bf16(ar0.v, b0.v, bd);
      bd = wmma_bf16(ar1.v, b1.v, bd);
#pragma unroll
      for (int v = 0; v < 8; ++v) bdl[(rbase + v) * 48 + nt * 16 + m] = bd[v];
    }

    // combine + causal mask + online softmax (per-row reductions in 16-lane halves)
#pragma unroll
    for (int v = 0; v < 8; ++v) {
      const int row = rbase + v;
      const int i = i0 + row;
      float x0 = (s0[v] + bdl[row * 48 + (15 - row + m)]) * scale;
      float x1 = (s1[v] + bdl[row * 48 + (31 - row + m)]) * scale;
      if (j0 + m > i) x0 = -1.0e30f;
      if (j0 + 16 + m > i) x1 = -1.0e30f;
      float mx = fmaxf(x0, x1);
#pragma unroll
      for (int off = 8; off >= 1; off >>= 1) mx = fmaxf(mx, __shfl_xor(mx, off, 32));
      float mnew = fmaxf(mrow[v], mx);
      float sf = __expf(mrow[v] - mnew);
      float e0 = __expf(x0 - mnew);
      float e1 = __expf(x1 - mnew);
      float rs = e0 + e1;
#pragma unroll
      for (int off = 8; off >= 1; off >>= 1) rs += __shfl_xor(rs, off, 32);
      lrow[v] = lrow[v] * sf + rs;
      mrow[v] = mnew;
      o0[v] *= sf; o1[v] *= sf; o2[v] *= sf; o3[v] *= sf;
      pl[row * 40 + m] = (bf16)e0;
      pl[row * 40 + 16 + m] = (bf16)e1;
    }

    // PV: O(16x64) += P(16x32) @ V(32x64); V read from transposed global layout
    FragA pa;
    pa.u[0] = *(const U128*)&pl[m * 40 + kh];
    pa.u[1] = *(const U128*)&pl[m * 40 + 16 + kh];
#pragma unroll
    for (int dt = 0; dt < 4; ++dt) {
      const bf16* vb = vtr + ((size_t)(bh * DHEAD + dt * 16 + m)) * SEQ_Q + j0;
      FragA bv;
      bv.u[0] = ldg16(vb + kh);
      bv.u[1] = ldg16(vb + 16 + kh);
      v8f& o = dt == 0 ? o0 : dt == 1 ? o1 : dt == 2 ? o2 : o3;
      o = wmma_bf16(pa.v, bv.v, o);
    }
  }

  // normalize and store attention vector (bf16 for the output GEMM)
#pragma unroll
  for (int v = 0; v < 8; ++v) {
    const int row = rbase + v;
    const float inv = 1.0f / lrow[v];
    bf16* dst = avec + ((size_t)(b * SEQ_Q + i0 + row) * DIM) + h * DHEAD + m;
    dst[0]  = (bf16)(o0[v] * inv);
    dst[16] = (bf16)(o1[v] * inv);
    dst[32] = (bf16)(o2[v] * inv);
    dst[48] = (bf16)(o3[v] * inv);
  }
}

// ---------------- residual + layernorm ----------------
__global__ __launch_bounds__(256) void resid_layernorm(
    const float* __restrict__ w, const float* __restrict__ ao,
    const float* __restrict__ gamma, const float* __restrict__ beta,
    float* __restrict__ out) {
  __shared__ float red[256];
  const int r = blockIdx.x;
  const int t = threadIdx.x;
  float x[4];
  float s = 0.f;
#pragma unroll
  for (int e = 0; e < 4; ++e) {
    int c = t + e * 256;
    x[e] = w[(size_t)r * DIM + c] + ao[(size_t)r * DIM + c];
    s += x[e];
  }
  red[t] = s;
  __syncthreads();
  for (int off = 128; off > 0; off >>= 1) {
    if (t < off) red[t] += red[t + off];
    __syncthreads();
  }
  const float mu = red[0] * (1.0f / DIM);
  __syncthreads();
  float vs = 0.f;
#pragma unroll
  for (int e = 0; e < 4; ++e) { float d = x[e] - mu; vs += d * d; }
  red[t] = vs;
  __syncthreads();
  for (int off = 128; off > 0; off >>= 1) {
    if (t < off) red[t] += red[t + off];
    __syncthreads();
  }
  const float rstd = rsqrtf(red[0] * (1.0f / DIM) + 1e-5f);
#pragma unroll
  for (int e = 0; e < 4; ++e) {
    int c = t + e * 256;
    out[(size_t)r * DIM + c] = (x[e] - mu) * rstd * gamma[c] + beta[c];
  }
}

extern "C" void kernel_launch(void* const* d_in, const int* in_sizes, int n_in,
                              void* d_out, int out_size, void* d_ws, size_t ws_size,
                              hipStream_t stream) {
  (void)in_sizes; (void)n_in; (void)out_size; (void)ws_size;
  const float* w     = (const float*)d_in[0];
  const float* r     = (const float*)d_in[1];
  const float* rwb   = (const float*)d_in[2];
  const float* rrb   = (const float*)d_in[3];
  const float* Wqkv  = (const float*)d_in[4];
  const float* Wr    = (const float*)d_in[5];
  const float* Wo    = (const float*)d_in[6];
  const float* gamma = (const float*)d_in[7];
  const float* beta  = (const float*)d_in[8];
  // d_in[9] = attn_mask: causality computed analytically, mask unused
  float* out = (float*)d_out;

  char* p = (char*)d_ws;
  auto carve = [&](size_t bytes) -> void* {
    void* q = (void*)p;
    p += (bytes + 255) & ~(size_t)255;
    return q;
  };
  bf16* w_bf    = (bf16*)carve((size_t)ROWS * DIM * 2);
  bf16* Wqkv_t  = (bf16*)carve((size_t)DIM * N_QKV * 2);  // (3072,1024) bf16
  bf16* r_bf    = (bf16*)carve((size_t)SEQ_Q * DIM * 2);
  bf16* Wr_t    = (bf16*)carve((size_t)DIM * DIM * 2);    // (1024,1024) bf16
  bf16* Wo_t    = (bf16*)carve((size_t)DIM * DIM * 2);
  bf16* qrw     = (bf16*)carve((size_t)ROWS * DIM * 2);
  bf16* qrr     = (bf16*)carve((size_t)ROWS * DIM * 2);
  bf16* kbf     = (bf16*)carve((size_t)ROWS * DIM * 2);
  bf16* vtr     = (bf16*)carve((size_t)ROWS * DIM * 2);   // V transposed per (b,h)
  bf16* rkb     = (bf16*)carve((size_t)SEQ_Q * DIM * 2);
  bf16* avec    = (bf16*)carve((size_t)ROWS * DIM * 2);
  float* aout   = (float*)carve((size_t)ROWS * DIM * 4);

  // activations: straight f32->bf16
  cvt_bf16<<<(ROWS * DIM + 255) / 256, 256, 0, stream>>>(w, w_bf, ROWS * DIM);
  cvt_bf16<<<(SEQ_Q * DIM + 255) / 256, 256, 0, stream>>>(r, r_bf, SEQ_Q * DIM);
  // weights: transpose-convert to (N,K) bf16 so GEMM staging is contiguous
  transpose_cvt_bf16<<<dim3(N_QKV / 32, DIM / 32), 256, 0, stream>>>(Wqkv, Wqkv_t,
                                                                     DIM, N_QKV);
  transpose_cvt_bf16<<<dim3(DIM / 32, DIM / 32), 256, 0, stream>>>(Wr, Wr_t, DIM,
                                                                   DIM);
  transpose_cvt_bf16<<<dim3(DIM / 32, DIM / 32), 256, 0, stream>>>(Wo, Wo_t, DIM,
                                                                   DIM);

  // QKV projection, fused bias-add + V-transpose epilogue
  gemm_wmma_bf16<<<dim3(N_QKV / 128, ROWS / 64), 256, 0, stream>>>(
      w_bf, Wqkv_t, ROWS, N_QKV, DIM, 0,
      qrw, qrr, kbf, vtr, rwb, rrb, nullptr, nullptr);

  // positional projection r @ W_r
  gemm_wmma_bf16<<<dim3(DIM / 128, SEQ_Q / 64), 256, 0, stream>>>(
      r_bf, Wr_t, SEQ_Q, DIM, DIM, 1,
      nullptr, nullptr, nullptr, nullptr, nullptr, nullptr, rkb, nullptr);

  // fused AC+BD flash attention
  attn_flash_wmma<<<dim3(SEQ_Q / 16, NUM_B * NHEAD), 32, 0, stream>>>(
      qrw, qrr, kbf, vtr, rkb, avec);

  // output projection
  gemm_wmma_bf16<<<dim3(DIM / 128, ROWS / 64), 256, 0, stream>>>(
      avec, Wo_t, ROWS, DIM, DIM, 2,
      nullptr, nullptr, nullptr, nullptr, nullptr, nullptr, nullptr, aout);

  // residual + layernorm
  resid_layernorm<<<ROWS, 256, 0, stream>>>(w, aout, gamma, beta, out);
}
